// EdgeQMul_68453188763805
// MI455X (gfx1250) — matrix operand order repaired and checked by reading, not compiled, not run on hardware
//
#include <hip/hip_runtime.h>
#include <stdint.h>

// Problem constants (match reference)
#define B_     128
#define N_     36
#define OBJ_   2048
#define QD_    1024
#define HID_   512
#define MID_   256
#define KOUT_  8
#define NN_    (N_*N_)            // 1296
#define EDGES_ (B_*N_*(N_-1))     // 161280

typedef __attribute__((ext_vector_type(16))) __bf16 v16bf;
typedef __attribute__((ext_vector_type(8)))  float  v8f;

union BF16x16 {
  v16bf v;
  unsigned short s[16];
  uint4 q[2];
};

// Native hardware conversion (v_cvt_*_bf16_f32) instead of software RNE.
__device__ __forceinline__ unsigned short f2bf(float x) {
  union { __bf16 b; unsigned short s; } t;
  t.b = (__bf16)x;
  return t.s;
}
__device__ __forceinline__ unsigned f2bf2(float lo, float hi) {
  union { unsigned short s[2]; unsigned u; } t;
  t.s[0] = f2bf(lo);
  t.s[1] = f2bf(hi);
  return t.u;
}

// ---------------------------------------------------------------------------
// Elementwise f32 -> bf16 (8 elements per thread, b128 in / b128 out).
// ---------------------------------------------------------------------------
__global__ void cvt_f32_bf16(const float* __restrict__ in,
                             unsigned short* __restrict__ outp, int n8) {
  int i = blockIdx.x * blockDim.x + threadIdx.x;
  if (i >= n8) return;
  const float4* p = (const float4*)in + (size_t)i * 2;
  float4 x = p[0], y = p[1];
  uint4 r;
  r.x = f2bf2(x.x, x.y);
  r.y = f2bf2(x.z, x.w);
  r.z = f2bf2(y.x, y.y);
  r.w = f2bf2(y.z, y.w);
  ((uint4*)outp)[i] = r;
}

// ---------------------------------------------------------------------------
// Weight-norm fold + bf16 convert + WMMA B-fragment packing.
// grid.x = number of rows (output cols), block = 256.
// ---------------------------------------------------------------------------
__global__ void pack_wn_bf16(const float* __restrict__ v,
                             const float* __restrict__ g,
                             unsigned short* __restrict__ outp,
                             int K, int ksteps) {
  __shared__ float red[256];
  const int r = blockIdx.x;
  const int tid = threadIdx.x;
  float s = 0.f;
  for (int k = tid; k < K; k += 256) {
    float x = v[(size_t)r * K + k];
    s += x * x;
  }
  red[tid] = s;
  __syncthreads();
  for (int off = 128; off > 0; off >>= 1) {
    if (tid < off) red[tid] += red[tid + off];
    __syncthreads();
  }
  const float scale = g[r] * rsqrtf(red[0]);
  const int nt = r >> 4, li = r & 15;
  for (int k = tid; k < K; k += 256) {
    int kk = k >> 5, ko = k & 31;
    int lane = li + (ko & 16);
    int elem = ko & 15;
    size_t oidx = ((((size_t)nt * ksteps) + kk) * 32 + lane) * 16 + elem;
    outp[oidx] = f2bf(v[(size_t)r * K + k] * scale);
  }
}

// Final-layer weights (8 x 256) packed as a zero-padded 16-column bf16
// B-fragment buffer (8 k-steps). grid.x = 16, block = 256.
__global__ void pack_w2_bf16(const float* __restrict__ v,
                             const float* __restrict__ g,
                             unsigned short* __restrict__ outp) {
  __shared__ float red[256];
  const int r = blockIdx.x;
  const int tid = threadIdx.x;
  float x = (r < KOUT_) ? v[(size_t)r * MID_ + tid] : 0.f;
  red[tid] = x * x;
  __syncthreads();
  for (int off = 128; off > 0; off >>= 1) {
    if (tid < off) red[tid] += red[tid + off];
    __syncthreads();
  }
  const float nrm = red[0];
  const float scale = (r < KOUT_ && nrm > 0.f) ? g[r] * rsqrtf(nrm) : 0.f;
  const int kk = tid >> 5, ko = tid & 31;
  const int lane = (r & 15) + (ko & 16);
  const int elem = ko & 15;
  size_t oidx = ((size_t)kk * 32 + lane) * 16 + elem;
  outp[oidx] = f2bf(x * scale);
}

// ---------------------------------------------------------------------------
// y = x @ Wᵀ + b, pre-packed bf16 W, pre-converted bf16 A, N fixed 512.
// 32 rows per workgroup (two 16-row A tiles share every B fragment).
// grid.x = M/32, block = 256 (8 waves, 64 cols each).
// ---------------------------------------------------------------------------
__global__ void __launch_bounds__(256)
wn_gemm_bf16(const unsigned short* __restrict__ Abf,
             const unsigned short* __restrict__ Wp,
             const float* __restrict__ bias,
             float* __restrict__ C,
             int K, int ksteps) {
  const int tid  = threadIdx.x;
  const int lane = tid & 31;
  const int wid  = tid >> 5;
  const int m0   = blockIdx.x * 32;
  const int n0   = wid * 64;

  const int row    = m0 + (lane & 15);
  const int kshift4 = (lane >> 4);           // 0/1 : uint4 shift within k-step
  const uint4* A0 = (const uint4*)(Abf + (size_t)row * K);
  const uint4* A1 = (const uint4*)(Abf + (size_t)(row + 16) * K);
  const uint4* W4 = (const uint4*)Wp;

  v8f c00 = {}, c01 = {}, c02 = {}, c03 = {};
  v8f c10 = {}, c11 = {}, c12 = {}, c13 = {};

  for (int kk = 0; kk < ksteps; ++kk) {
    const int ai = kk * 4 + kshift4 * 2;     // elements kb .. kb+7
    BF16x16 a0, a1;
    a0.q[0] = A0[ai];
    a0.q[1] = A0[ai + 1];                    // elements kb+8 .. kb+15
    a1.q[0] = A1[ai];
    a1.q[1] = A1[ai + 1];

#pragma unroll
    for (int nt = 0; nt < 4; ++nt) {
      const int ntg = (n0 >> 4) + nt;
      const int bi = (ntg * ksteps + kk) * 64 + lane * 2;
      BF16x16 b;
      b.q[0] = W4[bi];
      b.q[1] = W4[bi + 1];
      v8f* c0s[4] = { &c00, &c01, &c02, &c03 };
      v8f* c1s[4] = { &c10, &c11, &c12, &c13 };
      *c0s[nt] = __builtin_amdgcn_wmma_f32_16x16x32_bf16(
          false, a0.v, false, b.v, (short)0, *c0s[nt], false, false);
      *c1s[nt] = __builtin_amdgcn_wmma_f32_16x16x32_bf16(
          false, a1.v, false, b.v, (short)0, *c1s[nt], false, false);
    }
  }

  const int rb0 = m0 + (kshift4 << 3);       // C rows j / j+8 per half-wave
  const int rb1 = rb0 + 16;
  v8f c0s[4] = { c00, c01, c02, c03 };
  v8f c1s[4] = { c10, c11, c12, c13 };
#pragma unroll
  for (int nt = 0; nt < 4; ++nt) {
    const int col = n0 + nt * 16 + (lane & 15);
    const float bv = bias[col];
#pragma unroll
    for (int j = 0; j < 8; ++j) {
      C[(size_t)(rb0 + j) * HID_ + col] = c0s[nt][j] + bv;
      C[(size_t)(rb1 + j) * HID_ + col] = c1s[nt][j] + bv;
    }
  }
}

// ---------------------------------------------------------------------------
// Fused edge pipeline.  32 edges per workgroup.
//  phase 1: edge_feats = nf[i]*nf[j]*qp  -> bf16 LDS tile (32x512)
//  phase 2: WMMA 512->256 + bias + relu -> bf16 LDS tile (32x256)
//           each wave: 32-col strip x both 16-edge tiles (B reused 2x)
//  phase 3: WMMA 256->8 (zero-padded to 16 cols) + bias -> fp32 out
// ---------------------------------------------------------------------------
__global__ void __launch_bounds__(256)
edge_fused(const float* __restrict__ nf,
           const float* __restrict__ qp,
           const int* __restrict__ idx,
           const unsigned short* __restrict__ W1p,
           const float* __restrict__ b1,
           const unsigned short* __restrict__ W2p,
           const float* __restrict__ b2,
           float* __restrict__ out) {
  __shared__ unsigned short eT[32][520];   // edge feats, bf16, padded stride
  __shared__ unsigned short hT[32][264];   // hidden after relu, bf16

  const int tid  = threadIdx.x;
  const int lane = tid & 31;
  const int wid  = tid >> 5;
  const int e0   = blockIdx.x * 32;

  // ---- phase 1: build edge feature tile (8 threads per edge) ----
  {
    const int el  = tid >> 3;
    const int sub = tid & 7;
    const int gi  = idx[e0 + el];
    const int bi  = gi / NN_;
    const int rem = gi - bi * NN_;
    const int ii  = rem / N_;
    const int jj  = rem - ii * N_;
    const float* pi = nf + ((size_t)bi * N_ + ii) * HID_;
    const float* pj = nf + ((size_t)bi * N_ + jj) * HID_;
    const float* pq = qp + (size_t)bi * HID_;
#pragma unroll
    for (int c = 0; c < 16; ++c) {
      const int k = (c * 8 + sub) * 4;
      float4 a = *(const float4*)(pi + k);
      float4 b = *(const float4*)(pj + k);
      float4 q = *(const float4*)(pq + k);
      uint2 r;
      r.x = f2bf2(a.x * b.x * q.x, a.y * b.y * q.y);
      r.y = f2bf2(a.z * b.z * q.z, a.w * b.w * q.w);
      *(uint2*)&eT[el][k] = r;
    }
  }
  __syncthreads();

  // ---- phase 2: 512 -> 256 WMMA GEMM, bias + relu ----
  {
    const int rl0 = lane & 15;               // row in edge tile 0
    const int rl1 = 16 + rl0;                // row in edge tile 1
    const int kshift = (lane >> 4) << 3;
    const uint4* W4 = (const uint4*)W1p;

    v8f c00 = {}, c01 = {}, c10 = {}, c11 = {};   // c[mtile][nt]
#pragma unroll
    for (int kk = 0; kk < 16; ++kk) {
      const int kb = kk * 32 + kshift * 2;
      BF16x16 a0, a1;
      a0.q[0] = *(const uint4*)&eT[rl0][kb];
      a0.q[1] = *(const uint4*)&eT[rl0][kb + 8];
      a1.q[0] = *(const uint4*)&eT[rl1][kb];
      a1.q[1] = *(const uint4*)&eT[rl1][kb + 8];
      {
        const int bi = ((wid * 2 + 0) * 16 + kk) * 64 + lane * 2;
        BF16x16 b;
        b.q[0] = W4[bi];
        b.q[1] = W4[bi + 1];
        c00 = __builtin_amdgcn_wmma_f32_16x16x32_bf16(
            false, a0.v, false, b.v, (short)0, c00, false, false);
        c10 = __builtin_amdgcn_wmma_f32_16x16x32_bf16(
            false, a1.v, false, b.v, (short)0, c10, false, false);
      }
      {
        const int bi = ((wid * 2 + 1) * 16 + kk) * 64 + lane * 2;
        BF16x16 b;
        b.q[0] = W4[bi];
        b.q[1] = W4[bi + 1];
        c01 = __builtin_amdgcn_wmma_f32_16x16x32_bf16(
            false, a0.v, false, b.v, (short)0, c01, false, false);
        c11 = __builtin_amdgcn_wmma_f32_16x16x32_bf16(
            false, a1.v, false, b.v, (short)0, c11, false, false);
      }
    }

    const int rb0 = kshift;                  // rows j / j+8 of edge tile 0
    const int rb1 = 16 + kshift;             // rows of edge tile 1
#pragma unroll
    for (int nt = 0; nt < 2; ++nt) {
      const int col = wid * 32 + nt * 16 + (lane & 15);
      const float bv = b1[col];
      const v8f cA = nt ? c01 : c00;
      const v8f cB = nt ? c11 : c10;
#pragma unroll
      for (int j = 0; j < 8; ++j) {
        float h0 = cA[j] + bv;
        float h1 = cB[j] + bv;
        hT[rb0 + j][col] = f2bf(h0 > 0.f ? h0 : 0.f);
        hT[rb1 + j][col] = f2bf(h1 > 0.f ? h1 : 0.f);
      }
    }
  }
  __syncthreads();

  // ---- phase 3: 256 -> 8 on WMMA (waves 0..1, one 16-edge tile each) ----
  if (wid < 2) {
    const int rl = wid * 16 + (lane & 15);
    const int kshift = (lane >> 4) << 3;
    const uint4* W4 = (const uint4*)W2p;
    v8f acc = {};
#pragma unroll
    for (int kk = 0; kk < 8; ++kk) {         // K = 256 -> 8 k-steps
      const int kb = kk * 32 + kshift * 2;
      BF16x16 a;
      a.q[0] = *(const uint4*)&hT[rl][kb];
      a.q[1] = *(const uint4*)&hT[rl][kb + 8];
      const int bi = kk * 64 + lane * 2;
      BF16x16 b;
      b.q[0] = W4[bi];
      b.q[1] = W4[bi + 1];
      acc = __builtin_amdgcn_wmma_f32_16x16x32_bf16(
          false, a.v, false, b.v, (short)0, acc, false, false);
    }
    const int col = lane & 15;
    if (col < KOUT_) {                       // cols 8..15 are zero padding
      const float bv = b2[col];
      const int rb = wid * 16 + kshift;
#pragma unroll
      for (int j = 0; j < 8; ++j)
        out[(size_t)(e0 + rb + j) * KOUT_ + col] = acc[j] + bv;
    }
  }
}

// ---------------------------------------------------------------------------
extern "C" void kernel_launch(void* const* d_in, const int* in_sizes, int n_in,
                              void* d_out, int out_size, void* d_ws, size_t ws_size,
                              hipStream_t stream) {
  const float* node  = (const float*)d_in[0];   // (B, N, OBJ)
  const float* qf    = (const float*)d_in[1];   // (B, QD)
  const int*   idx   = (const int*)d_in[2];     // (EDGES,)
  const float* v_obj = (const float*)d_in[3];
  const float* g_obj = (const float*)d_in[4];
  const float* b_obj = (const float*)d_in[5];
  const float* v_q   = (const float*)d_in[6];
  const float* g_q   = (const float*)d_in[7];
  const float* b_q   = (const float*)d_in[8];
  const float* v_l1  = (const float*)d_in[9];
  const float* g_l1  = (const float*)d_in[10];
  const float* b_l1  = (const float*)d_in[11];
  const float* v_l2  = (const float*)d_in[12];
  const float* g_l2  = (const float*)d_in[13];
  const float* b_l2  = (const float*)d_in[14];

  char* ws = (char*)d_ws;
  unsigned short* Wobj   = (unsigned short*)(ws + 0);         // 2 MB
  unsigned short* Wq     = (unsigned short*)(ws + 2097152);   // 1 MB
  unsigned short* Wl1    = (unsigned short*)(ws + 3145728);   // 256 KB
  unsigned short* W2p    = (unsigned short*)(ws + 3407872);   // 8 KB
  float*          nf     = (float*)(ws + 3416064);            // 9.4 MB
  float*          qp     = (float*)(ws + 12853248);           // 256 KB
  unsigned short* nodebf = (unsigned short*)(ws + 13115392);  // 18.9 MB
  unsigned short* qbf    = (unsigned short*)(ws + 31989760);  // 256 KB

  // 0) pre-convert GEMM A-operands to bf16 (bandwidth-bound, tiny cost)
  cvt_f32_bf16<<<(B_ * N_ * OBJ_ / 8 + 255) / 256, 256, 0, stream>>>(
      node, nodebf, B_ * N_ * OBJ_ / 8);
  cvt_f32_bf16<<<(B_ * QD_ / 8 + 255) / 256, 256, 0, stream>>>(
      qf, qbf, B_ * QD_ / 8);

  // 1) fold weight-norm, convert to bf16, pack to WMMA B-fragment layout
  pack_wn_bf16<<<HID_, 256, 0, stream>>>(v_obj, g_obj, Wobj, OBJ_, OBJ_ / 32);
  pack_wn_bf16<<<HID_, 256, 0, stream>>>(v_q,   g_q,   Wq,   QD_,  QD_ / 32);
  pack_wn_bf16<<<MID_, 256, 0, stream>>>(v_l1,  g_l1,  Wl1,  HID_, HID_ / 32);
  pack_w2_bf16<<<16, 256, 0, stream>>>(v_l2, g_l2, W2p);

  // 2) nf = node_feats @ W_objᵀ + b_obj   (4608 x 512)
  wn_gemm_bf16<<<(B_ * N_) / 32, 256, 0, stream>>>(nodebf, Wobj, b_obj, nf,
                                                   OBJ_, OBJ_ / 32);
  // 3) qp = q_feats @ W_qᵀ + b_q          (128 x 512)
  wn_gemm_bf16<<<B_ / 32, 256, 0, stream>>>(qbf, Wq, b_q, qp, QD_, QD_ / 32);

  // 4) fused edge pipeline -> out (161280 x 8)
  edge_fused<<<EDGES_ / 32, 256, 0, stream>>>(nf, qp, idx, Wl1, b_l1, W2p, b_l2,
                                              (float*)d_out);
}
